// MultiHeadSelfAttention_395136991786
// MI455X (gfx1250) — matrix operand order repaired and checked by reading, not compiled, not run on hardware
//
#include <hip/hip_runtime.h>

typedef __attribute__((ext_vector_type(16))) __bf16 v16bf;
typedef __attribute__((ext_vector_type(8)))  float  v8f;

#define HEADS  16
#define BATCH  8
#define SEQLEN 512
#define INSZ   1024
#define OUTSZ  64
#define INV_TEMP (1.0f / 32.0f)   // 1/sqrt(1024)

static __device__ __forceinline__ unsigned short f32_to_bf16(float f) {
  union { float f; unsigned u; } x; x.f = f;
  unsigned r = x.u + 0x7FFFu + ((x.u >> 16) & 1u);  // round-to-nearest-even
  return (unsigned short)(r >> 16);
}

// A-fragment K index for 16-bit 16x32 A layout: lanes 0-15 hold K {0..7,16..23},
// lanes 16-31 hold K {8..15,24..31}; pairs packed per VGPR.
static __device__ __forceinline__ int a_frag_k(int j, int half) {
  return ((j < 4) ? (2 * j) : (16 + 2 * (j - 4))) + 8 * half;
}

// CDNA5 async global->LDS copy, 16B per lane, tracked by ASYNCcnt.
static __device__ __forceinline__ void async_copy_b128(void* lds_ptr, const void* gptr) {
  unsigned lds_off = (unsigned)(size_t)lds_ptr;   // flat LDS addr low 32b = LDS offset
  asm volatile("global_load_async_to_lds_b128 %0, %1, off"
               :: "v"(lds_off), "v"(gptr) : "memory");
}
static __device__ __forceinline__ void async_wait0() {
  asm volatile("s_wait_asynccnt 0x0" ::: "memory");
}

// ---------------------------------------------------------------------------
// Kernel 0a: one-time f32 -> bf16 of activations: kq = seq + pos, sv = seq.
// 4 contiguous elements per thread (b128 loads, packed stores).
// ---------------------------------------------------------------------------
__global__ void __launch_bounds__(256)
convert_seq_kernel(const float* __restrict__ seq,
                   const float* __restrict__ pos,
                   unsigned short* __restrict__ kq,
                   unsigned short* __restrict__ sv)
{
  const size_t i4 = ((size_t)blockIdx.x * 256 + threadIdx.x) * 4;
  const size_t p4 = i4 & ((size_t)SEQLEN * INSZ - 1);   // pos repeats per batch
  float4 s = *(const float4*)(seq + i4);
  float4 p = *(const float4*)(pos + p4);
  unsigned kqp[2], svp[2];
  kqp[0] = (unsigned)f32_to_bf16(s.x + p.x) | ((unsigned)f32_to_bf16(s.y + p.y) << 16);
  kqp[1] = (unsigned)f32_to_bf16(s.z + p.z) | ((unsigned)f32_to_bf16(s.w + p.w) << 16);
  svp[0] = (unsigned)f32_to_bf16(s.x) | ((unsigned)f32_to_bf16(s.y) << 16);
  svp[1] = (unsigned)f32_to_bf16(s.z) | ((unsigned)f32_to_bf16(s.w) << 16);
  *(uint2*)(kq + i4) = make_uint2(kqp[0], kqp[1]);
  *(uint2*)(sv + i4) = make_uint2(svp[0], svp[1]);
}

// ---------------------------------------------------------------------------
// Kernel 0b: one-time weight transpose + bf16: WT[(proj,h,n), k] = W[proj][h][k][n]
// ---------------------------------------------------------------------------
__global__ void __launch_bounds__(256)
convert_w_kernel(const float* __restrict__ Wq,
                 const float* __restrict__ Wk,
                 const float* __restrict__ Wv,
                 unsigned short* __restrict__ WT)
{
  const unsigned o = blockIdx.x * 256 + threadIdx.x;   // < 3*16*64*1024
  const unsigned proj = o >> 20;
  const unsigned h    = (o >> 16) & 15;
  const unsigned n    = (o >> 10) & 63;
  const unsigned k    = o & 1023;
  const float* W = (proj == 0) ? Wq : (proj == 1) ? Wk : Wv;
  WT[o] = f32_to_bf16(W[((size_t)h * INSZ + k) * OUTSZ + n]);
}

// ---------------------------------------------------------------------------
// Kernel 1: Q/K/V projections, all-bf16 operands, async LDS staging.
// grid = (4096/64, HEADS, 3)  block = 128 (4 waves; wave -> 16-row strip)
// K-chunk = 64 -> 8 WMMAs per chunk per wave, 16 chunks.
// ---------------------------------------------------------------------------
__global__ void __launch_bounds__(128)
proj_gemm_kernel(const unsigned short* __restrict__ kq,
                 const unsigned short* __restrict__ sv,
                 const unsigned short* __restrict__ WT,
                 unsigned short* __restrict__ pq,
                 unsigned short* __restrict__ pk,
                 unsigned short* __restrict__ pvT)
{
  __shared__ unsigned short ldsA[64 * 64];   // [row][k] bf16 (8 KB)
  __shared__ unsigned short ldsB[64 * 64];   // [n][k]   bf16 (8 KB)

  const int tid  = threadIdx.x;
  const int wave = tid >> 5;
  const int lane = tid & 31;
  const int half = lane >> 4;
  const int nl   = lane & 15;

  const int m0   = blockIdx.x * 64;
  const int h    = blockIdx.y;
  const int proj = blockIdx.z;           // 0=q, 1=k, 2=v

  const unsigned short* X   = (proj == 2) ? sv : kq;
  const unsigned short* WTh = WT + ((size_t)proj * HEADS + h) * OUTSZ * INSZ;

  v8f acc[4] = {};

  for (int k0 = 0; k0 < INSZ; k0 += 64) {
    // Async-stage A tile 64x64 bf16 (rows of X) and B tile [n][k] 64x64 bf16.
    #pragma unroll
    for (int i = 0; i < 4; ++i) {
      int e = tid + i * 128;            // 512 x 16B segments
      int row = e >> 3, seg = e & 7;    // 8 segments of 8 bf16 per row
      async_copy_b128(&ldsA[row * 64 + seg * 8],
                      X + (size_t)(m0 + row) * INSZ + k0 + seg * 8);
      async_copy_b128(&ldsB[row * 64 + seg * 8],
                      WTh + (size_t)row * INSZ + k0 + seg * 8);
    }
    async_wait0();
    __syncthreads();

    union { v16bf v; unsigned u[8]; } fa[2];
    #pragma unroll
    for (int cc = 0; cc < 2; ++cc)
      #pragma unroll
      for (int j = 0; j < 8; ++j)
        fa[cc].u[j] = *(const unsigned*)&ldsA[(wave * 16 + nl) * 64 + cc * 32 + a_frag_k(j, half)];

    #pragma unroll
    for (int t = 0; t < 4; ++t) {
      #pragma unroll
      for (int cc = 0; cc < 2; ++cc) {
        union { v16bf v; unsigned u[8]; } fb;
        #pragma unroll
        for (int j = 0; j < 8; ++j)   // B: lane=n, K pairs {0..7,16..23}/{8..15,24..31} per half
          fb.u[j] = *(const unsigned*)&ldsB[(t * 16 + nl) * 64 + cc * 32 + a_frag_k(j, half)];
        acc[t] = __builtin_amdgcn_wmma_f32_16x16x32_bf16(
            false, fa[cc].v, false, fb.v, (short)0, acc[t], false, false);
      }
    }
    __syncthreads();
  }

  // Store accumulators (C layout: row = r + 8*half, col = 16*t + nl)
  #pragma unroll
  for (int t = 0; t < 4; ++t) {
    #pragma unroll
    for (int r = 0; r < 8; ++r) {
      int M = m0 + wave * 16 + r + 8 * half;
      int b = M >> 9;
      int l = M & (SEQLEN - 1);
      int d = t * 16 + nl;
      unsigned short v = f32_to_bf16(acc[t][r]);
      size_t bh = (size_t)b * HEADS + h;
      if (proj == 0)      pq [(bh * SEQLEN + l) * OUTSZ + d] = v;
      else if (proj == 1) pk [(bh * SEQLEN + l) * OUTSZ + d] = v;
      else                pvT[(bh * OUTSZ + d) * SEQLEN + l] = v;   // transposed
    }
  }
}

// ---------------------------------------------------------------------------
// Kernel 2: flash attention.  Per wave: one 16-query tile; loop 32-key blocks.
// grid = (SEQLEN/64, HEADS, BATCH)  block = 128, shared K/V staging (async).
// ---------------------------------------------------------------------------
__global__ void __launch_bounds__(128)
attention_kernel(const unsigned short* __restrict__ pq,
                 const unsigned short* __restrict__ pk,
                 const unsigned short* __restrict__ pvT,
                 const float* __restrict__ mask,
                 float* __restrict__ out)
{
  __shared__ unsigned short ldsK[32 * 64];     // [key][d]
  __shared__ unsigned short ldsVt[64 * 32];    // [feat][key]
  __shared__ unsigned short ldsP[4][16 * 32];  // per-wave P tile (C->A relayout)

  const int tid  = threadIdx.x;
  const int wave = tid >> 5;
  const int lane = tid & 31;
  const int half = lane >> 4;
  const int nl   = lane & 15;

  const int h = blockIdx.y;
  const int b = blockIdx.z;
  const size_t bh = (size_t)b * HEADS + h;
  const int q0 = (blockIdx.x * 4 + wave) * 16;

  union Frag { v16bf v; unsigned u[8]; };

  // Q fragments: 16x64 -> two 16x32 bf16 A fragments, kept in registers.
  Frag fq[2];
  const unsigned short* qbase = pq + (bh * SEQLEN + q0) * OUTSZ;
  #pragma unroll
  for (int c = 0; c < 2; ++c)
    #pragma unroll
    for (int j = 0; j < 8; ++j)
      fq[c].u[j] = *(const unsigned*)&qbase[(size_t)nl * OUTSZ + c * 32 + a_frag_k(j, half)];

  v8f acc[4] = {};
  float rowm[8], rowl[8];
  #pragma unroll
  for (int r = 0; r < 8; ++r) { rowm[r] = -__builtin_inff(); rowl[r] = 0.0f; }

  const unsigned short* kbase = pk  + bh * SEQLEN * OUTSZ;
  const unsigned short* vbase = pvT + bh * OUTSZ * SEQLEN;

  for (int kb = 0; kb < SEQLEN / 32; ++kb) {
    // Async-stage 32 keys of K (4 KB contiguous) and Vt (64 rows x 32 keys).
    #pragma unroll
    for (int i = 0; i < 2; ++i) {
      int e = tid + i * 128;            // 256 x 16B segments
      async_copy_b128((unsigned short*)ldsK + e * 8,
                      kbase + (size_t)kb * 32 * OUTSZ + e * 8);
      int f = e >> 2, c = e & 3;        // 4 segments per feat row
      async_copy_b128(&ldsVt[f * 32 + c * 8],
                      vbase + (size_t)f * SEQLEN + kb * 32 + c * 8);
    }
    async_wait0();
    __syncthreads();

    // S tiles: two 16x16 tiles, each = 2 WMMAs over d (K=64 contraction)
    v8f s[2];
    #pragma unroll
    for (int kt = 0; kt < 2; ++kt) {
      v8f c0 = {};
      #pragma unroll
      for (int cc = 0; cc < 2; ++cc) {
        Frag fb;   // B(k=d, n=key): contiguous d pairs in ldsK row 'key'
        #pragma unroll
        for (int j = 0; j < 8; ++j)
          fb.u[j] = *(const unsigned*)&ldsK[(kt * 16 + nl) * 64 + cc * 32 + a_frag_k(j, half)];
        c0 = __builtin_amdgcn_wmma_f32_16x16x32_bf16(
            false, fq[cc].v, false, fb.v, (short)0, c0, false, false);
      }
      s[kt] = c0;
    }

    float km0 = mask[(size_t)b * SEQLEN + kb * 32 + nl];
    float km1 = mask[(size_t)b * SEQLEN + kb * 32 + 16 + nl];

    unsigned short* ldsPw = ldsP[wave];
    #pragma unroll
    for (int r = 0; r < 8; ++r) {
      int row = r + 8 * half;
      float v0 = s[0][r] * INV_TEMP;
      float v1 = s[1][r] * INV_TEMP;
      if (km0 == 0.0f) v0 = -__builtin_inff();
      if (km1 == 0.0f) v1 = -__builtin_inff();
      float vm = fmaxf(v0, v1);
      #pragma unroll
      for (int off = 8; off >= 1; off >>= 1)   // reduce over N (16 lanes/half)
        vm = fmaxf(vm, __shfl_xor(vm, off, 16));
      float newm = fmaxf(rowm[r], vm);
      float corr = __expf(rowm[r] - newm);
      rowm[r] = newm;
      float p0 = __expf(v0 - newm);
      float p1 = __expf(v1 - newm);
      float ps = p0 + p1;
      #pragma unroll
      for (int off = 8; off >= 1; off >>= 1)
        ps += __shfl_xor(ps, off, 16);
      rowl[r] = rowl[r] * corr + ps;
      #pragma unroll
      for (int t = 0; t < 4; ++t) acc[t][r] *= corr;   // rescale running O
      ldsPw[row * 32 + nl]      = f32_to_bf16(p0);
      ldsPw[row * 32 + 16 + nl] = f32_to_bf16(p1);
    }
    asm volatile("s_wait_dscnt 0" ::: "memory");  // P stores -> A-frag loads

    Frag fp;
    #pragma unroll
    for (int j = 0; j < 8; ++j)
      fp.u[j] = *(const unsigned*)&ldsPw[nl * 32 + a_frag_k(j, half)];

    #pragma unroll
    for (int ft = 0; ft < 4; ++ft) {
      Frag fv;   // B(k=key, n=feat): contiguous key pairs in ldsVt row 'feat'
      #pragma unroll
      for (int j = 0; j < 8; ++j)
        fv.u[j] = *(const unsigned*)&ldsVt[(ft * 16 + nl) * 32 + a_frag_k(j, half)];
      acc[ft] = __builtin_amdgcn_wmma_f32_16x16x32_bf16(
          false, fp.v, false, fv.v, (short)0, acc[ft], false, false);
    }
    __syncthreads();
  }

  // Finalize: divide by row sums, apply query mask, write [B,Q,H,V] f32.
  #pragma unroll
  for (int r = 0; r < 8; ++r) {
    int row = r + 8 * half;
    int q = q0 + row;
    float qm  = mask[(size_t)b * SEQLEN + q];
    float inv = (rowl[r] > 0.0f && qm != 0.0f) ? (1.0f / rowl[r]) : 0.0f;
    #pragma unroll
    for (int ft = 0; ft < 4; ++ft)
      out[(((size_t)b * SEQLEN + q) * HEADS + h) * OUTSZ + ft * 16 + nl] = acc[ft][r] * inv;
  }
}

extern "C" void kernel_launch(void* const* d_in, const int* in_sizes, int n_in,
                              void* d_out, int out_size, void* d_ws, size_t ws_size,
                              hipStream_t stream) {
  const float* seq  = (const float*)d_in[0];
  const float* mask = (const float*)d_in[1];
  const float* pos  = (const float*)d_in[2];
  const float* Wq   = (const float*)d_in[3];
  const float* Wk   = (const float*)d_in[4];
  const float* Wv   = (const float*)d_in[5];
  float* out = (float*)d_out;

  const size_t nAct = (size_t)BATCH * SEQLEN * INSZ;            // 4 Mi
  const size_t nW   = (size_t)3 * HEADS * OUTSZ * INSZ;         // 3 Mi
  const size_t nPrj = (size_t)BATCH * HEADS * SEQLEN * OUTSZ;   // 4 Mi

  unsigned short* kq  = (unsigned short*)d_ws;
  unsigned short* sv  = kq + nAct;
  unsigned short* WT  = sv + nAct;
  unsigned short* pq  = WT + nW;
  unsigned short* pk  = pq + nPrj;
  unsigned short* pvT = pk + nPrj;

  convert_seq_kernel<<<dim3(nAct / 4 / 256), 256, 0, stream>>>(seq, pos, kq, sv);
  convert_w_kernel<<<dim3(nW / 256), 256, 0, stream>>>(Wq, Wk, Wv, WT);

  dim3 g1(BATCH * SEQLEN / 64, HEADS, 3);
  proj_gemm_kernel<<<g1, 128, 0, stream>>>(kq, sv, WT, pq, pk, pvT);

  dim3 g2(SEQLEN / 64, HEADS, BATCH);
  attention_kernel<<<g2, 128, 0, stream>>>(pq, pk, pvT, mask, out);
}
